// ProductBasisLayer_10144712753176
// MI455X (gfx1250) — matrix operand order repaired and checked by reading, not compiled, not run on hardware
//
#include <hip/hip_runtime.h>

// ---------------------------------------------------------------------------
// Problem constants (match the JAX reference)
// ---------------------------------------------------------------------------
#define NN 16384   // nodes
#define FF 512     // features
#define SS 10      // species
#define MT 32      // nodes per block (two 16-row WMMA M-subtiles)
// C = 3 correlations (hard-coded / unrolled below)

typedef __attribute__((ext_vector_type(16))) __bf16 v16bf;
typedef __attribute__((ext_vector_type(8)))  float  v8f;

struct Pack32 { int4 a, b; };  // 32 bytes == one v16bf fragment

// LDS A-tile layout: [kb][mt][ch][row][kw]   (k split into 8 blocks of 64)
//  - row stride padded to 72 shorts (144 B = 36 dwords; gcd(36,64)=4 ->
//    16 distinct start banks for the 16 fragment rows, conflict-free b128)
//  - all offsets within one k-block body fit the 16-bit ds immediate field
#define ROW_STRIDE 72                    // shorts (64 data + 8 pad)
#define CH_STRIDE  (16 * ROW_STRIDE)     // 1152 shorts  (2304 B)
#define MT_STRIDE  (4 * CH_STRIDE)       // 4608 shorts  (9216 B)
#define KB_STRIDE  (2 * MT_STRIDE)       // 9216 shorts  (18432 B)
#define NKB        (FF / 64)             // 8 k-blocks
// total LDS = NKB * KB_STRIDE shorts = 144 KB

static __device__ __forceinline__ unsigned short f2bf(float f) {
  // round-to-nearest-even f32 -> bf16
  unsigned int u = __float_as_uint(f);
  unsigned int r = u + 0x7FFFu + ((u >> 16) & 1u);
  return (unsigned short)(r >> 16);
}

// A-fragment (16x32 bf16, row M = lane%16):
//  lanes 0-15 : K = [kw, kw+8) then [kw+16, kw+24)
//  lanes 16-31: K = [kw+8, kw+16) then [kw+24, kw+32)
static __device__ __forceinline__ v16bf load_a_frag(const unsigned short* row,
                                                    int kw, int hi) {
  Pack32 p;
  p.a = *(const int4*)(row + kw + hi * 8);
  p.b = *(const int4*)(row + kw + 16 + hi * 8);
  return __builtin_bit_cast(v16bf, p);
}

// B-fragment (32x16 bf16, col N = lane%16):
//  lanes 0-15 : K = [k0, k0+8) then [k0+16, k0+24)
//  lanes 16-31: K = [k0+8, k0+16) then [k0+24, k0+32)
//  wt is stored transposed: wt[chan][g][k] -> two 16-byte chunks per lane.
static __device__ __forceinline__ v16bf load_b_frag(const unsigned short* row,
                                                    int k0, int hi) {
  Pack32 q;
  q.a = *(const int4*)(row + k0 + hi * 8);
  q.b = *(const int4*)(row + k0 + 16 + hi * 8);
  return __builtin_bit_cast(v16bf, q);
}

// ---------------------------------------------------------------------------
// Pre-kernel 1: convert W_s / W_v (f32 row-major [f][g]) into bf16 transposed
//               layout wt[chan][g][k] so WMMA B fragments are contiguous.
// ---------------------------------------------------------------------------
__global__ void prep_weights_kernel(const float* __restrict__ Ws,
                                    const float* __restrict__ Wv,
                                    unsigned short* __restrict__ wt) {
  int idx = blockIdx.x * blockDim.x + threadIdx.x;
  if (idx >= 2 * FF * FF) return;
  int c = idx / (FF * FF);
  int r = idx - c * (FF * FF);
  int g = r / FF;     // output column
  int k = r - g * FF; // contraction index
  const float* W = c ? Wv : Ws;
  wt[idx] = f2bf(W[k * FF + g]);
}

// ---------------------------------------------------------------------------
// Pre-kernel 2: species[n] = argmax(node_attrs[n, :])  (undo the one-hot)
// ---------------------------------------------------------------------------
__global__ void prep_species_kernel(const float* __restrict__ attrs,
                                    int* __restrict__ spec) {
  int n = blockIdx.x * blockDim.x + threadIdx.x;
  if (n >= NN) return;
  int s = 0;
  float best = attrs[n * SS];
  for (int k = 1; k < SS; ++k) {
    float v = attrs[n * SS + k];
    if (v > best) { best = v; s = k; }
  }
  spec[n] = s;
}

// ---------------------------------------------------------------------------
// Fused kernel: per-block 32-node tile.
//   Phase 1: product-basis recurrence -> bf16 A tiles in LDS.
//   Phase 2: 4 bf16 WMMA GEMMs (scalar @ W_s, 3 vector comps @ W_v) over two
//            16-row M-subtiles sharing each B fragment (8 WMMAs per B load),
//            f32 accumulate, fused +sc epilogue, float4 stores.
// ---------------------------------------------------------------------------
__global__ __launch_bounds__(256) void fused_product_basis_kernel(
    const float* __restrict__ node_feats,  // (N, F, 4)
    const float* __restrict__ sc,          // (N, F, 4)
    const float* __restrict__ w_prod,      // (2, 4, S, F)
    const float* __restrict__ w_out,       // (3, 2, S, F)
    const unsigned short* __restrict__ wt, // bf16 (2, F, F) transposed weights
    const int* __restrict__ spec,          // (N,)
    float* __restrict__ out)               // (N, F, 4)
{
  // 144 KB of LDS: A tiles, [kb][mt][ch][row][kw] bf16 (padded rows)
  __shared__ unsigned short Abuf[NKB * KB_STRIDE];

  const int tid = threadIdx.x;
  const int nodeBase = blockIdx.x * MT;

  // ---------------- Phase 1: elementwise recurrence ----------------
  #pragma unroll 4
  for (int i = 0; i < (MT * FF) / 256; ++i) {
    int p = tid + i * 256;
    int r = p >> 9;          // local node row (0..MT-1)
    int f = p & (FF - 1);    // feature
    int node = nodeBase + r;
    int s = spec[node];

    float4 x = *(const float4*)(node_feats + (size_t)(node * FF + f) * 4);
    const float xs = x.x, xv0 = x.y, xv1 = x.z, xv2 = x.w;

    float wp[2][4];
    #pragma unroll
    for (int c = 0; c < 2; ++c)
      #pragma unroll
      for (int q = 0; q < 4; ++q)
        wp[c][q] = w_prod[((c * 4 + q) * SS + s) * FF + f];

    float wo[3][2];
    #pragma unroll
    for (int c = 0; c < 3; ++c)
      #pragma unroll
      for (int l = 0; l < 2; ++l)
        wo[c][l] = w_out[((c * 2 + l) * SS + s) * FF + f];

    float Ts = xs, Tx = xv0, Ty = xv1, Tz = xv2;
    float os = wo[0][0] * Ts;
    float ox = wo[0][1] * Tx;
    float oy = wo[0][1] * Ty;
    float oz = wo[0][1] * Tz;

    #pragma unroll
    for (int c = 1; c < 3; ++c) {
      float dot = Tx * xv0 + Ty * xv1 + Tz * xv2;
      float ns = wp[c - 1][0] * Ts * xs + wp[c - 1][1] * dot;
      float nx = wp[c - 1][2] * Ts * xv0 + wp[c - 1][3] * Tx * xs;
      float ny = wp[c - 1][2] * Ts * xv1 + wp[c - 1][3] * Ty * xs;
      float nz = wp[c - 1][2] * Ts * xv2 + wp[c - 1][3] * Tz * xs;
      Ts = ns; Tx = nx; Ty = ny; Tz = nz;
      os += wo[c][0] * Ts;
      ox += wo[c][1] * Tx;
      oy += wo[c][1] * Ty;
      oz += wo[c][1] * Tz;
    }

    // [kb][mt][ch][row][kw] layout
    int mt  = r >> 4;
    int row = r & 15;
    int kb  = f >> 6;
    int kw  = f & 63;
    unsigned short* dst =
        &Abuf[kb * KB_STRIDE + mt * MT_STRIDE + row * ROW_STRIDE + kw];
    dst[0 * CH_STRIDE] = f2bf(os);
    dst[1 * CH_STRIDE] = f2bf(ox);
    dst[2 * CH_STRIDE] = f2bf(oy);
    dst[3 * CH_STRIDE] = f2bf(oz);
  }

  __syncthreads();

  // ---------------- Phase 2: WMMA GEMMs ----------------
  const int wave = tid >> 5;
  const int lane = tid & 31;
  const int l16  = lane & 15;
  const int hi   = lane >> 4;

  // Single per-lane base; all per-body tile offsets are <16.5 KB immediates.
  const unsigned short* abase = &Abuf[l16 * ROW_STRIDE];

  for (int gt = wave; gt < FF / 16; gt += 8) {
    const int g = gt * 16 + l16;                 // output column for this lane
    const unsigned short* brow_s = wt + (size_t)g * FF;            // W_s^T row
    const unsigned short* brow_v = wt + (size_t)(FF + g) * FF;     // W_v^T row

    v8f accS[2] = {{}, {}}, accX[2] = {{}, {}};
    v8f accY[2] = {{}, {}}, accZ[2] = {{}, {}};

    #pragma unroll 2
    for (int k0 = 0; k0 < FF; k0 += 32) {
      const int kb = k0 >> 6;
      const int kw = k0 & 63;
      const unsigned short* akb = abase + kb * KB_STRIDE;

      v16bf bS = load_b_frag(brow_s, k0, hi);
      v16bf bV = load_b_frag(brow_v, k0, hi);

      #pragma unroll
      for (int mt = 0; mt < 2; ++mt) {
        const unsigned short* ab = akb + mt * MT_STRIDE;
        v16bf aS = load_a_frag(ab + 0 * CH_STRIDE, kw, hi);
        v16bf aX = load_a_frag(ab + 1 * CH_STRIDE, kw, hi);
        v16bf aY = load_a_frag(ab + 2 * CH_STRIDE, kw, hi);
        v16bf aZ = load_a_frag(ab + 3 * CH_STRIDE, kw, hi);

        accS[mt] = __builtin_amdgcn_wmma_f32_16x16x32_bf16(
            false, aS, false, bS, (short)0, accS[mt], false, false);
        accX[mt] = __builtin_amdgcn_wmma_f32_16x16x32_bf16(
            false, aX, false, bV, (short)0, accX[mt], false, false);
        accY[mt] = __builtin_amdgcn_wmma_f32_16x16x32_bf16(
            false, aY, false, bV, (short)0, accY[mt], false, false);
        accZ[mt] = __builtin_amdgcn_wmma_f32_16x16x32_bf16(
            false, aZ, false, bV, (short)0, accZ[mt], false, false);
      }
    }

    // Epilogue: D layout -> VGPR j holds row M = j + 8*hi, column N = lane%16.
    #pragma unroll
    for (int mt = 0; mt < 2; ++mt) {
      #pragma unroll
      for (int j = 0; j < 8; ++j) {
        int m = mt * 16 + j + hi * 8;
        int node = nodeBase + m;
        size_t off = (size_t)(node * FF + g) * 4;
        float4 scv = *(const float4*)(sc + off);
        float4 rv;
        rv.x = accS[mt][j] + scv.x;
        rv.y = accX[mt][j] + scv.y;
        rv.z = accY[mt][j] + scv.z;
        rv.w = accZ[mt][j] + scv.w;
        *(float4*)(out + off) = rv;
      }
    }
  }
}

// ---------------------------------------------------------------------------
// Host launcher
// ---------------------------------------------------------------------------
extern "C" void kernel_launch(void* const* d_in, const int* in_sizes, int n_in,
                              void* d_out, int out_size, void* d_ws, size_t ws_size,
                              hipStream_t stream) {
  const float* node_feats = (const float*)d_in[0]; // (N, F, 4)
  const float* sc         = (const float*)d_in[1]; // (N, F, 4)
  const float* node_attrs = (const float*)d_in[2]; // (N, S)
  const float* w_prod     = (const float*)d_in[3]; // (2, 4, S, F)
  const float* w_out      = (const float*)d_in[4]; // (3, 2, S, F)
  const float* W_s        = (const float*)d_in[5]; // (F, F)
  const float* W_v        = (const float*)d_in[6]; // (F, F)

  // Workspace layout: [0, 1 MB) bf16 transposed weights; then species ints.
  unsigned short* wt = (unsigned short*)d_ws;
  int* spec = (int*)((char*)d_ws + (size_t)2 * FF * FF * sizeof(unsigned short));

  prep_weights_kernel<<<(2 * FF * FF + 255) / 256, 256, 0, stream>>>(W_s, W_v, wt);
  prep_species_kernel<<<(NN + 255) / 256, 256, 0, stream>>>(node_attrs, spec);

  fused_product_basis_kernel<<<NN / MT, 256, 0, stream>>>(
      node_feats, sc, w_prod, w_out, wt, spec, (float*)d_out);
}
